// Neighbor_55748675502371
// MI455X (gfx1250) — matrix operand order repaired and checked by reading, not compiled, not run on hardware
//
#include <hip/hip_runtime.h>
#include <hip/hip_bf16.h>
#include <stdint.h>
#include <stddef.h>

// Problem constants (fixed by reference setup_inputs()).
#define N_PTS   8192
#define DIM     512
#define DPAD    520    // 512 bf16 + 8 bf16 (16B) LDS pad per row (TDM pad_amount)
#define TOPK    16
#define NRUNS   5
#define NCENT   64
#define KMITERS 20

typedef __attribute__((ext_vector_type(16))) __bf16 bf16x16;
typedef __attribute__((ext_vector_type(8)))  __bf16 bf16x8;
typedef __attribute__((ext_vector_type(8)))  float  f32x8;
typedef __attribute__((ext_vector_type(4)))  unsigned int u32x4;
typedef __attribute__((ext_vector_type(4)))  int i32x4;
typedef __attribute__((ext_vector_type(8)))  int i32x8;

__device__ __forceinline__ bf16x16 join16(bf16x8 lo, bf16x8 hi) {
  return __builtin_shufflevector(lo, hi, 0,1,2,3,4,5,6,7,8,9,10,11,12,13,14,15);
}

// A fragment (16x32 bf16): lane&15 = row M; lanes 0-15 hold
// K = [0..15], lanes 16-31 hold K = [16..31] (matches B fragment).
__device__ __forceinline__ bf16x16 load_a_frag(const __bf16* M, int rowBase, int lane, int kstep) {
  const int r    = lane & 15;
  const int koff = (lane >> 4) * 16;
  const __bf16* p = M + (size_t)(rowBase + r) * DIM + kstep * 32;
  bf16x8 lo = *reinterpret_cast<const bf16x8*>(p + koff);
  bf16x8 hi = *reinterpret_cast<const bf16x8*>(p + 8 + koff);
  return join16(lo, hi);
}

// ---- Tensor Data Mover: load nrows_tile rows of DIM bf16 (row stride DIM)
// into LDS with 16B pad after each 1024B row (-> DPAD bf16 LDS row stride).
// D# layout per CDNA5 ISA.
__device__ __forceinline__ void tdm_load_rows(unsigned lds_off, const void* gsrc,
                                              int nrows_tensor, int nrows_tile) {
  const unsigned long long ga = (unsigned long long)(uintptr_t)gsrc;
  u32x4 g0 = {};
  g0[0] = 1u;                                               // count=1 (valid)
  g0[1] = lds_off;                                          // lds_addr (bytes)
  g0[2] = (unsigned)ga;                                     // global_addr[31:0]
  g0[3] = (unsigned)((ga >> 32) & 0x1FFFFFFu) | (2u << 30); // [56:32] | type=2
  i32x8 g1 = {};
  // data_size=2B (1<<16), pad_enable (1<<20), pad_interval=256dw (7<<22),
  // pad_amount=4dw (3<<25)
  g1[0] = (int)((1u << 16) | (1u << 20) | (7u << 22) | (3u << 25));
  g1[1] = (int)((unsigned)(DIM & 0xffff) << 16);            // tensor_dim0[15:0]
  g1[2] = (int)(((unsigned)nrows_tensor & 0xffff) << 16);   // tensor_dim1[15:0]
  g1[3] = (int)((unsigned)(DIM & 0xffff) << 16);            // tile_dim0 = 512
  g1[4] = (int)((unsigned)nrows_tile & 0xffff);             // tile_dim1
  g1[5] = DIM;                                              // tensor_dim0_stride
  i32x4 gz = {};
#if defined(__clang_major__) && __clang_major__ >= 23
  i32x8 gz8 = {};
  __builtin_amdgcn_tensor_load_to_lds(g0, g1, gz, gz, gz8, 0);
#else
  __builtin_amdgcn_tensor_load_to_lds(g0, g1, gz, gz, 0);
#endif
}

// ---------------------------------------------------------------- convert
__global__ void cvt_bf16_kernel(const float* __restrict__ src, __bf16* __restrict__ dst, int n) {
  int i = blockIdx.x * blockDim.x + threadIdx.x;
  const int stride = gridDim.x * blockDim.x;
  for (; i < n; i += stride) dst[i] = (__bf16)src[i];
}

// ----------------------------------------------- fused sim GEMM + top-16
// 8 waves / block, each wave owns 16 rows and sweeps all 8192 columns in
// 16-wide WMMA tiles. The B chunk (16 cols x 512 K, 16KB) is TDM-staged into
// LDS once per block (double buffered) instead of 8x through the vmem path.
__global__ __launch_bounds__(256) void simtopk_kernel(const __bf16* __restrict__ S,
                                                      const __bf16* __restrict__ T,
                                                      int* __restrict__ knn) {
  __shared__ __bf16 s_b[2][16][DPAD];      // ~32.5 KB double-buffered B tile
  __shared__ float s_tile[8][16][16];
  __shared__ float s_topv[8][16][16];
  __shared__ int   s_topc[8][16][16];
  __shared__ float s_minv[8][16];
  __shared__ int   s_minp[8][16];

  const int wave    = threadIdx.x >> 5;
  const int lane    = threadIdx.x & 31;
  const int rowBase = blockIdx.x * 128 + wave * 16;

  // Preload the full K=512 A stripe into registers: 16 fragments = 128 VGPRs.
  bf16x16 a[16];
  #pragma unroll
  for (int k = 0; k < 16; ++k) a[k] = load_a_frag(S, rowBase, lane, k);

  if (lane < 16) {
    #pragma unroll
    for (int j = 0; j < TOPK; ++j) { s_topv[wave][lane][j] = -__builtin_inff(); s_topc[wave][lane][j] = 0; }
    s_minv[wave][lane] = -__builtin_inff();
    s_minp[wave][lane] = 0;
  }

  if (wave == 0) {                         // prologue: stage chunk 0
    tdm_load_rows((unsigned)(size_t)(void*)&s_b[0][0][0], T, N_PTS, 16);
    __builtin_amdgcn_s_wait_tensorcnt(0);
  }
  __syncthreads();

  const int NCH = N_PTS / 16;
  for (int chunk = 0; chunk < NCH; ++chunk) {
    const int cur     = chunk & 1;
    const int colBase = chunk * 16;
    if (wave == 0 && chunk + 1 < NCH)      // async-stage next chunk (TENSORcnt)
      tdm_load_rows((unsigned)(size_t)(void*)&s_b[cur ^ 1][0][0],
                    T + (size_t)(chunk + 1) * 16 * DIM, N_PTS, 16);

    f32x8 acc = {};
    #pragma unroll
    for (int k = 0; k < 16; ++k) {
      // B fragment from LDS: lane&15 = col, 16 contiguous K bf16 per lane.
      const bf16x8* bp =
          reinterpret_cast<const bf16x8*>(&s_b[cur][lane & 15][k * 32 + (lane >> 4) * 16]);
      bf16x16 b = join16(bp[0], bp[1]);
      acc = __builtin_amdgcn_wmma_f32_16x16x32_bf16(false, a[k], false, b, (short)0, acc, false, false);
    }

    const int colg = colBase + (lane & 15);
    #pragma unroll
    for (int v = 0; v < 8; ++v) {
      const int rowg = rowBase + v + (lane >> 4) * 8;
      float val = acc[v] + ((rowg == colg) ? 10.0f : 0.0f);  // +10*eye
      s_tile[wave][v + (lane >> 4) * 8][lane & 15] = val;
    }
    __syncthreads();

    if (lane < 16) {                       // top-16 with threshold test
      const int r = lane;
      float curMin = s_minv[wave][r];
      int   minPos = s_minp[wave][r];
      for (int cc = 0; cc < 16; ++cc) {
        const float val = s_tile[wave][r][cc];
        if (val > curMin) {                // rare insert path (~100/row total)
          s_topv[wave][r][minPos] = val;
          s_topc[wave][r][minPos] = colBase + cc;
          float m = s_topv[wave][r][0]; int mp = 0;
          #pragma unroll
          for (int t = 1; t < TOPK; ++t) {
            const float tv = s_topv[wave][r][t];
            if (tv < m) { m = tv; mp = t; }
          }
          curMin = m; minPos = mp;
        }
      }
      s_minv[wave][r] = curMin; s_minp[wave][r] = minPos;
    }
    if (wave == 0 && chunk + 1 < NCH)
      __builtin_amdgcn_s_wait_tensorcnt(0); // next buffer resident
    __syncthreads();                        // publish to all waves
  }

  if (lane < 16) {
    #pragma unroll
    for (int j = 0; j < TOPK; ++j)
      knn[(size_t)(rowBase + lane) * TOPK + j] = s_topc[wave][lane][j];
  }
}

// -------------------------------------- k-means centroid init / finalize
__global__ __launch_bounds__(128) void cent_update_kernel(
    const float* __restrict__ teacher, float* __restrict__ cent,
    __bf16* __restrict__ cent_bf, float* __restrict__ cnorm,
    float* __restrict__ sums, int* __restrict__ counts, int mode) {
  const int run = blockIdx.x / NCENT;
  const int c   = blockIdx.x % NCENT;
  const int tid = threadIdx.x;
  float*  myc  = cent    + ((size_t)run * NCENT + c) * DIM;
  __bf16* mycb = cent_bf + ((size_t)run * NCENT + c) * DIM;
  float*  mys  = sums    + ((size_t)run * NCENT + c) * DIM;
  __shared__ float red[128];

  float part = 0.f;
  if (mode == 1) {
    // Deterministic seeding (stand-in for jax Threefry choice): injective in c.
    const int idx = (127 * c + 1301 * run + 11) & (N_PTS - 1);
    for (int d = tid; d < DIM; d += 128) {
      const float v = teacher[(size_t)idx * DIM + d];
      myc[d] = v; mycb[d] = (__bf16)v; mys[d] = 0.f;
      part += v * v;
    }
  } else {
    const int cnt   = counts[run * NCENT + c];
    const float inv = (cnt > 0) ? (1.0f / (float)cnt) : 0.f;
    for (int d = tid; d < DIM; d += 128) {
      const float v = (cnt > 0) ? mys[d] * inv : myc[d];
      myc[d] = v; mycb[d] = (__bf16)v; mys[d] = 0.f;
      part += v * v;
    }
  }
  if (tid == 0) counts[run * NCENT + c] = 0;

  red[tid] = part;
  __syncthreads();
  for (int s = 64; s > 0; s >>= 1) {
    if (tid < s) red[tid] += red[tid + s];
    __syncthreads();
  }
  if (tid == 0) cnorm[run * NCENT + c] = red[0];
}

// ------------------------------------------------ k-means assign (WMMA)
// dist = ||c||^2 - 2 x.c ; one wave computes 16 rows x 64 centroids.
// Centroids are TDM-staged into LDS 32 rows at a time.
__global__ __launch_bounds__(32) void assign_kernel(
    const __bf16* __restrict__ Xbf, const __bf16* __restrict__ cent_bf,
    const float* __restrict__ cnorm, int* __restrict__ labels) {
  __shared__ __bf16 s_c[32][DPAD];   // ~32.5 KB centroid staging
  __shared__ float s_d[16][NCENT];
  const int lane    = threadIdx.x & 31;
  const int rowBase = blockIdx.x * 16;
  const int run     = blockIdx.y;
  const __bf16* C  = cent_bf + (size_t)run * NCENT * DIM;
  const float*  NN = cnorm + run * NCENT;

  bf16x16 a[16];
  #pragma unroll
  for (int k = 0; k < 16; ++k) a[k] = load_a_frag(Xbf, rowBase, lane, k);

  for (int half = 0; half < 2; ++half) {
    if (half)  // WAR guard: previous ds reads of s_c must retire before TDM rewrite
      asm volatile("s_wait_dscnt 0" ::: "memory");
    tdm_load_rows((unsigned)(size_t)(void*)&s_c[0][0],
                  C + (size_t)half * 32 * DIM, NCENT, 32);
    __builtin_amdgcn_s_wait_tensorcnt(0);

    #pragma unroll
    for (int ct2 = 0; ct2 < 2; ++ct2) {
      f32x8 acc = {};
      #pragma unroll
      for (int k = 0; k < 16; ++k) {
        const bf16x8* bp = reinterpret_cast<const bf16x8*>(
            &s_c[ct2 * 16 + (lane & 15)][k * 32 + (lane >> 4) * 16]);
        bf16x16 b = join16(bp[0], bp[1]);
        acc = __builtin_amdgcn_wmma_f32_16x16x32_bf16(false, a[k], false, b, (short)0, acc, false, false);
      }
      const int cg = half * 32 + ct2 * 16 + (lane & 15);
      const float cn = NN[cg];
      #pragma unroll
      for (int v = 0; v < 8; ++v)
        s_d[v + (lane >> 4) * 8][cg] = cn - 2.0f * acc[v];
    }
  }
  __syncthreads();

  if (lane < 16) {
    float best = s_d[lane][0]; int bi = 0;
    for (int c = 1; c < NCENT; ++c) {
      const float d = s_d[lane][c];
      if (d < best) { best = d; bi = c; }   // strict <: first index on ties
    }
    labels[(size_t)run * N_PTS + rowBase + lane] = bi;
  }
}

// ------------------------------------------------ segment-sum via atomics
__global__ __launch_bounds__(256) void seg_sum_kernel(
    const float* __restrict__ X, const int* __restrict__ labels,
    float* __restrict__ sums, int* __restrict__ counts) {
  const int run     = blockIdx.y;
  const int wave    = threadIdx.x >> 5;
  const int lane    = threadIdx.x & 31;
  const int rowBase = blockIdx.x * 64 + wave * 8;
  for (int rr = 0; rr < 8; ++rr) {
    const int row = rowBase + rr;
    const int lab = labels[(size_t)run * N_PTS + row];
    float* dst       = sums + ((size_t)run * NCENT + lab) * DIM;
    const float* src = X + (size_t)row * DIM;
    if (rr + 1 < 8)  // gfx1250 global_prefetch_b8 of next row
      __builtin_prefetch(src + DIM + lane * 16, 0, 1);
    #pragma unroll
    for (int k = 0; k < DIM / 32; ++k) {
      const int d = lane + k * 32;
      atomicAdd(&dst[d], src[d]);
    }
    if (lane == 0) atomicAdd(&counts[run * NCENT + lab], 1);
  }
}

// ------------------------------------------------------- sparse scatter
// pos[i, knn[i,j]] = adj[i, knn[i,j]] + (any_run labels match ? 1 : 0)
__global__ __launch_bounds__(256) void scatter_kernel(
    const float* __restrict__ adj, const int* __restrict__ knn,
    const int* __restrict__ labels, float* __restrict__ out) {
  const int tid = threadIdx.x;
  const int row = blockIdx.x * 16 + (tid >> 4);
  const int j   = tid & 15;
  const int c   = knn[(size_t)row * TOPK + j];
  int m = 0;
  #pragma unroll
  for (int r = 0; r < NRUNS; ++r)
    m |= (labels[(size_t)r * N_PTS + row] == labels[(size_t)r * N_PTS + c]);
  out[(size_t)row * N_PTS + c] = adj[(size_t)row * N_PTS + c] + (m ? 1.0f : 0.0f);
}

// ----------------------------------------------------------------- driver
extern "C" void kernel_launch(void* const* d_in, const int* in_sizes, int n_in,
                              void* d_out, int out_size, void* d_ws, size_t ws_size,
                              hipStream_t stream) {
  (void)in_sizes; (void)n_in; (void)out_size; (void)ws_size;
  const float* adj     = (const float*)d_in[0];
  const float* student = (const float*)d_in[1];
  const float* teacher = (const float*)d_in[2];
  // d_in[3] = topk (==16), baked in as TOPK.

  char* ws = (char*)d_ws;
  size_t cur = 0;
  auto alloc = [&](size_t bytes) -> void* {
    void* p = ws + cur;
    cur = (cur + bytes + 255) & ~(size_t)255;
    return p;
  };
  __bf16* s_bf    = (__bf16*)alloc((size_t)N_PTS * DIM * 2);
  __bf16* t_bf    = (__bf16*)alloc((size_t)N_PTS * DIM * 2);
  int*    knn     = (int*)   alloc((size_t)N_PTS * TOPK * 4);
  int*    labels  = (int*)   alloc((size_t)NRUNS * N_PTS * 4);
  float*  cent    = (float*) alloc((size_t)NRUNS * NCENT * DIM * 4);
  __bf16* cent_bf = (__bf16*)alloc((size_t)NRUNS * NCENT * DIM * 2);
  float*  cnorm   = (float*) alloc((size_t)NRUNS * NCENT * 4);
  float*  sums    = (float*) alloc((size_t)NRUNS * NCENT * DIM * 4);
  int*    counts  = (int*)   alloc((size_t)NRUNS * NCENT * 4);
  // total ~19 MB of workspace

  cvt_bf16_kernel<<<2048, 256, 0, stream>>>(student, s_bf, N_PTS * DIM);
  cvt_bf16_kernel<<<2048, 256, 0, stream>>>(teacher, t_bf, N_PTS * DIM);

  simtopk_kernel<<<N_PTS / 128, 256, 0, stream>>>(s_bf, t_bf, knn);

  cent_update_kernel<<<NRUNS * NCENT, 128, 0, stream>>>(teacher, cent, cent_bf, cnorm, sums, counts, 1);
  for (int it = 0; it < KMITERS; ++it) {
    assign_kernel<<<dim3(N_PTS / 16, NRUNS), 32, 0, stream>>>(t_bf, cent_bf, cnorm, labels);
    seg_sum_kernel<<<dim3(N_PTS / 64, NRUNS), 256, 0, stream>>>(teacher, labels, sums, counts);
    cent_update_kernel<<<NRUNS * NCENT, 128, 0, stream>>>(teacher, cent, cent_bf, cnorm, sums, counts, 0);
  }
  assign_kernel<<<dim3(N_PTS / 16, NRUNS), 32, 0, stream>>>(t_bf, cent_bf, cnorm, labels);

  hipMemsetAsync(d_out, 0, (size_t)N_PTS * N_PTS * sizeof(float), stream);
  scatter_kernel<<<N_PTS / 16, 256, 0, stream>>>(adj, knn, labels, (float*)d_out);
}
